// Attention_111669150032
// MI455X (gfx1250) — matrix operand order repaired and checked by reading, not compiled, not run on hardware
//
#include <hip/hip_runtime.h>
#include <hip/hip_bf16.h>
#include <math.h>

typedef _Float16 v16h __attribute__((ext_vector_type(16)));
typedef float    v8f  __attribute__((ext_vector_type(8)));
typedef int      v4i  __attribute__((ext_vector_type(4)));

#define Bv   2
#define Sv   2048
#define DIMv 1024
#define NHv  16
#define HDv  64

#if defined(__has_builtin)
#if __has_builtin(__builtin_amdgcn_global_load_async_to_lds_b128)
#define HAVE_ASYNC_LDS 1
#endif
#endif

typedef __attribute__((address_space(1))) v4i gv4i_t;   // global int4
typedef __attribute__((address_space(3))) v4i lv4i_t;   // LDS int4

union Frag16 { v16h v; _Float16 h[16]; unsigned u[8]; };

// ---- A fragment: row-major 16-bit A (16x32), lda in halfs -------------------
// lane row = lane&15 ; group g = lane>>4 ; element e -> k = e + g*16
__device__ __forceinline__ v16h load_a_rowmajor(const _Float16* A, int lda,
                                                int m0, int k0, int lane) {
    int row = m0 + (lane & 15);
    int g   = lane >> 4;
    const unsigned* p0 = (const unsigned*)(A + (size_t)row * lda + k0 + g * 16);
    const unsigned* p1 = (const unsigned*)(A + (size_t)row * lda + k0 + g * 16 + 8);
    Frag16 f;
    f.u[0] = p0[0]; f.u[1] = p0[1]; f.u[2] = p0[2]; f.u[3] = p0[3];
    f.u[4] = p1[0]; f.u[5] = p1[1]; f.u[6] = p1[2]; f.u[7] = p1[3];
    return f.v;
}

// ---- B fragment where B[k,n] = W[n,k], W row-major with stride ldw ----------
// column = lane&15 ; element e -> k = e + (lane>>4)*16  (contiguous in k)
__device__ __forceinline__ v16h load_b_colk(const _Float16* W, int ldw,
                                            int n0, int k0, int lane) {
    int col = n0 + (lane & 15);
    int g   = lane >> 4;
    const unsigned* p = (const unsigned*)(W + (size_t)col * ldw + k0 + g * 16);
    Frag16 f;
#pragma unroll
    for (int t = 0; t < 8; ++t) f.u[t] = p[t];
    return f.v;
}

// ---- B fragment from a 32x16 16-bit tile in LDS (row-major, stride 16) ------
__device__ __forceinline__ v16h load_b_ldstile(const _Float16* T, int lane) {
    int col = lane & 15;
    int g   = lane >> 4;
    Frag16 f;
#pragma unroll
    for (int e = 0; e < 16; ++e)
        f.h[e] = T[(e + g * 16) * 16 + col];
    return f.v;
}

// ---- A fragment from LDS f32 scores (row-major, stride lds), convert to f16 -
__device__ __forceinline__ v16h load_a_lds_f32(const float* S, int lds,
                                               int k0, int lane) {
    int row = lane & 15;
    int g   = lane >> 4;
    Frag16 f;
#pragma unroll
    for (int e = 0; e < 16; ++e) {
        int k = k0 + e + g * 16;
        f.h[e] = (_Float16)S[row * lds + k];
    }
    return f.v;
}

// ============================================================================
// Kernel 0: f32 -> f16 conversion
// ============================================================================
__global__ void cvt_f16_kernel(const float* __restrict__ src,
                               _Float16* __restrict__ dst, int n) {
    for (int i = blockIdx.x * blockDim.x + threadIdx.x; i < n;
         i += gridDim.x * blockDim.x)
        dst[i] = (_Float16)src[i];
}

// ============================================================================
// Kernel 1: fused QKV projection + RoPE, 4x N-blocked (wave -> 16x64 tile).
// grid = (4, 256, 3) ; block = 128 (4 waves).
// z = 0:Q(+rope) 1:K(+rope) 2:V. Output layout [b, h, s, d] in f16.
// ============================================================================
__global__ void qkv_rope_kernel(const _Float16* __restrict__ xh,
                                const _Float16* __restrict__ wqh,
                                const _Float16* __restrict__ wkh,
                                const _Float16* __restrict__ wvh,
                                const float* __restrict__ fcos,
                                const float* __restrict__ fsin,
                                _Float16* __restrict__ qo,
                                _Float16* __restrict__ ko,
                                _Float16* __restrict__ vo) {
    int wave  = threadIdx.x >> 5;
    int lane  = threadIdx.x & 31;
    int nbase = (blockIdx.x * 4 + wave) * 64;   // wave covers 64 output cols
    int m0    = blockIdx.y * 16;
    int z     = blockIdx.z;

    const _Float16* W = (z == 0) ? wqh : (z == 1) ? wkh : wvh;
    _Float16*       O = (z == 0) ? qo  : (z == 1) ? ko  : vo;

    v8f acc[4] = {{}, {}, {}, {}};
#pragma unroll 2
    for (int k0 = 0; k0 < DIMv; k0 += 32) {
        v16h a = load_a_rowmajor(xh, DIMv, m0, k0, lane);   // reused 4x
#pragma unroll
        for (int t = 0; t < 4; ++t) {
            v16h b = load_b_colk(W, DIMv, nbase + t * 16, k0, lane);
            acc[t] = __builtin_amdgcn_wmma_f32_16x16x32_f16(false, a, false, b,
                                                            (short)0, acc[t],
                                                            false, false);
        }
    }

    int g   = lane >> 4;
    int col = lane & 15;

#pragma unroll
    for (int t = 0; t < 4; ++t) {
        int n    = nbase + t * 16 + col;
        int pidx = (n & (HDv - 1)) >> 1;     // RoPE pair index within head

        if (z < 2) { // apply RoPE to Q and K
#pragma unroll
            for (int j = 0; j < 8; ++j) {
                int   m    = m0 + j + g * 8;
                int   spos = m & (Sv - 1);
                float cv   = fcos[spos * (HDv / 2) + pidx];
                float sv   = fsin[spos * (HDv / 2) + pidx];
                float own  = acc[t][j];
                float prt  = __shfl_xor(own, 1, 32);
                acc[t][j]  = (n & 1) ? (prt * sv + own * cv)   // odd
                                     : (own * cv - prt * sv);  // even
            }
        }
#pragma unroll
        for (int j = 0; j < 8; ++j) {
            int m = m0 + j + g * 8;
            int b_ = m >> 11, s = m & (Sv - 1);
            int h_ = n >> 6,  d = n & (HDv - 1);
            O[(((size_t)(b_ * NHv + h_) * Sv + s) * HDv) + d] = (_Float16)acc[t][j];
        }
    }
}

// ============================================================================
// Kernel 2: attention per (b, h, 16-row q-tile). 4 waves, 132KB LDS.
//   phase 1: scores = (q kT)/8 -> LDS (A frags hoisted, WMMA)
//   phase 2: softmax in LDS, write attn tile to d_out (single 536MB stream)
//   phase 3: out = attn @ V  — V staged via async global->LDS copies
// grid = (128, 16, 2) ; block = 128 ; dyn LDS = 16*2048*4 + 4*32*16*2 bytes
// ============================================================================
__global__ void attn_kernel(const _Float16* __restrict__ qh,
                            const _Float16* __restrict__ kh,
                            const _Float16* __restrict__ vh,
                            float* __restrict__ attn_out,
                            _Float16* __restrict__ aoh) {
    extern __shared__ float sc[];                   // [16][2048] scores
    _Float16* vstage = (_Float16*)(sc + 16 * Sv);   // [4][32][16] V staging

    const int wave = threadIdx.x >> 5;
    const int lane = threadIdx.x & 31;
    const int q0   = blockIdx.x * 16;
    const int h_   = blockIdx.y;
    const int b_   = blockIdx.z;

    const size_t headoff = (size_t)(b_ * NHv + h_) * Sv * HDv;
    const _Float16* qp = qh + headoff;
    const _Float16* kp = kh + headoff;
    const _Float16* vp = vh + headoff;

    // ---- phase 1: scores -> LDS (Q fragments loop-invariant) ---------------
    {
        v16h a0 = load_a_rowmajor(qp, HDv, q0, 0,  lane);
        v16h a1 = load_a_rowmajor(qp, HDv, q0, 32, lane);
        for (int nt = wave; nt < Sv / 16; nt += 4) {
            v8f acc = {};
            v16h b0 = load_b_colk(kp, HDv, nt * 16, 0, lane);
            acc = __builtin_amdgcn_wmma_f32_16x16x32_f16(false, a0, false, b0,
                                                         (short)0, acc, false, false);
            v16h b1 = load_b_colk(kp, HDv, nt * 16, 32, lane);
            acc = __builtin_amdgcn_wmma_f32_16x16x32_f16(false, a1, false, b1,
                                                         (short)0, acc, false, false);
            int g = lane >> 4, col = nt * 16 + (lane & 15);
#pragma unroll
            for (int j = 0; j < 8; ++j)
                sc[(j + g * 8) * Sv + col] = acc[j] * 0.125f;   // 1/sqrt(64)
        }
    }
    __syncthreads();

    // ---- phase 2: softmax + stream normalized weights to d_out -------------
#pragma unroll
    for (int r = 0; r < 4; ++r) {
        int row = wave * 4 + r;
        float* rp = sc + row * Sv;

        float mx = -INFINITY;
        for (int c = lane; c < Sv; c += 32) mx = fmaxf(mx, rp[c]);
#pragma unroll
        for (int off = 16; off > 0; off >>= 1)
            mx = fmaxf(mx, __shfl_xor(mx, off, 32));

        float sum = 0.f;
        for (int c = lane; c < Sv; c += 32) {
            float e = __expf(rp[c] - mx);
            rp[c] = e;
            sum += e;
        }
#pragma unroll
        for (int off = 16; off > 0; off >>= 1)
            sum += __shfl_xor(sum, off, 32);

        float inv = 1.f / sum;
        size_t base = ((size_t)(b_ * NHv + h_) * Sv + (q0 + row)) * Sv;
        for (int c = lane; c < Sv; c += 32) {
            float w = rp[c] * inv;
            rp[c] = w;
            attn_out[base + c] = w;
        }
    }
    __syncthreads();

    // ---- phase 3: out = attn @ V  (wave -> 16 cols; V staged in LDS) -------
    {
        int n0 = wave * 16;
        _Float16* vtile = vstage + wave * (32 * 16);   // this wave's 32x16 tile
        v8f acc = {};
        for (int k0 = 0; k0 < Sv; k0 += 32) {
            // stage V rows [k0, k0+32) cols [n0, n0+16): lane -> one 32B row
            const _Float16* gsrc = vp + (size_t)(k0 + lane) * HDv + n0;
            _Float16*       ldst = vtile + lane * 16;
#ifdef HAVE_ASYNC_LDS
            gv4i_t* g0 = (gv4i_t*)(unsigned long long)(const char*)gsrc;
            lv4i_t* l0 = (lv4i_t*)(unsigned int)(unsigned long long)(char*)ldst;
            __builtin_amdgcn_global_load_async_to_lds_b128(g0, l0, 0, 0);
            gv4i_t* g1 = (gv4i_t*)(unsigned long long)(const char*)(gsrc + 8);
            lv4i_t* l1 = (lv4i_t*)(unsigned int)(unsigned long long)(char*)(ldst + 8);
            __builtin_amdgcn_global_load_async_to_lds_b128(g1, l1, 0, 0);
            asm volatile("s_wait_asynccnt 0x0" ::: "memory");
#else
            const unsigned* gs = (const unsigned*)gsrc;
            unsigned*       ls = (unsigned*)ldst;
#pragma unroll
            for (int t = 0; t < 8; ++t) ls[t] = gs[t];
#endif
            v16h a = load_a_lds_f32(sc, Sv, k0, lane);
            v16h b = load_b_ldstile(vtile, lane);
            acc = __builtin_amdgcn_wmma_f32_16x16x32_f16(false, a, false, b,
                                                         (short)0, acc, false, false);
        }
        int g = lane >> 4, d = n0 + (lane & 15);
#pragma unroll
        for (int j = 0; j < 8; ++j) {
            int s = q0 + j + g * 8;
            aoh[(((size_t)(b_ * Sv + s)) * NHv + h_) * HDv + d] = (_Float16)acc[j];
        }
    }
}

// ============================================================================
// Kernel 3: output projection  out = attn_out @ wo^T  (f32), 4x N-blocked
// grid = (4, 256) ; block = 128
// ============================================================================
__global__ void oproj_kernel(const _Float16* __restrict__ aoh,
                             const _Float16* __restrict__ woh,
                             float* __restrict__ out) {
    int wave  = threadIdx.x >> 5;
    int lane  = threadIdx.x & 31;
    int nbase = (blockIdx.x * 4 + wave) * 64;
    int m0    = blockIdx.y * 16;

    v8f acc[4] = {{}, {}, {}, {}};
#pragma unroll 2
    for (int k0 = 0; k0 < DIMv; k0 += 32) {
        v16h a = load_a_rowmajor(aoh, DIMv, m0, k0, lane);
#pragma unroll
        for (int t = 0; t < 4; ++t) {
            v16h b = load_b_colk(woh, DIMv, nbase + t * 16, k0, lane);
            acc[t] = __builtin_amdgcn_wmma_f32_16x16x32_f16(false, a, false, b,
                                                            (short)0, acc[t],
                                                            false, false);
        }
    }
    int g = lane >> 4, col = lane & 15;
#pragma unroll
    for (int t = 0; t < 4; ++t) {
        int n = nbase + t * 16 + col;
#pragma unroll
        for (int j = 0; j < 8; ++j) {
            int m = m0 + j + g * 8;
            out[(size_t)m * DIMv + n] = acc[t][j];
        }
    }
}

// ============================================================================
extern "C" void kernel_launch(void* const* d_in, const int* in_sizes, int n_in,
                              void* d_out, int out_size, void* d_ws, size_t ws_size,
                              hipStream_t stream) {
    (void)in_sizes; (void)n_in; (void)out_size; (void)ws_size;

    const float* x    = (const float*)d_in[0];
    const float* wq   = (const float*)d_in[1];
    const float* wk   = (const float*)d_in[2];
    const float* wv   = (const float*)d_in[3];
    const float* wo   = (const float*)d_in[4];
    const float* fcos = (const float*)d_in[5];
    const float* fsin = (const float*)d_in[6];

    float* out      = (float*)d_out;                       // (B,S,DIM)
    float* attn_out = out + (size_t)Bv * Sv * DIMv;        // (B,NH,S,S)

    // workspace layout (halfs)
    _Float16* wsh = (_Float16*)d_ws;
    const size_t NXD = (size_t)Bv * Sv * DIMv;  // 4,194,304
    const size_t NW  = (size_t)DIMv * DIMv;     // 1,048,576
    _Float16* xh  = wsh;
    _Float16* wqh = xh  + NXD;
    _Float16* wkh = wqh + NW;
    _Float16* wvh = wkh + NW;
    _Float16* woh = wvh + NW;
    _Float16* qh  = woh + NW;
    _Float16* kh  = qh  + NXD;
    _Float16* vh  = kh  + NXD;
    _Float16* aoh = vh  + NXD;

    // 0) f32 -> f16 conversions
    cvt_f16_kernel<<<dim3(2048), dim3(256), 0, stream>>>(x,  xh,  (int)NXD);
    cvt_f16_kernel<<<dim3(512),  dim3(256), 0, stream>>>(wq, wqh, (int)NW);
    cvt_f16_kernel<<<dim3(512),  dim3(256), 0, stream>>>(wk, wkh, (int)NW);
    cvt_f16_kernel<<<dim3(512),  dim3(256), 0, stream>>>(wv, wvh, (int)NW);
    cvt_f16_kernel<<<dim3(512),  dim3(256), 0, stream>>>(wo, woh, (int)NW);

    // 1) QKV projection + RoPE (wave -> 16x64 tile, A-frag reuse x4)
    qkv_rope_kernel<<<dim3(DIMv / 256, (Bv * Sv) / 16, 3), dim3(128), 0, stream>>>(
        xh, wqh, wkh, wvh, fcos, fsin, qh, kh, vh);

    // 2) attention (128KB scores + 4KB V staging per WG; 2 WG per 320KB WGP)
    const int lds_bytes = 16 * Sv * (int)sizeof(float) + 4 * 32 * 16 * (int)sizeof(_Float16);
    (void)hipFuncSetAttribute((const void*)attn_kernel,
                              hipFuncAttributeMaxDynamicSharedMemorySize, lds_bytes);
    attn_kernel<<<dim3(Sv / 16, NHv, Bv), dim3(128), lds_bytes, stream>>>(
        qh, kh, vh, attn_out, aoh);

    // 3) output projection
    oproj_kernel<<<dim3(DIMv / 256, (Bv * Sv) / 16), dim3(128), 0, stream>>>(
        aoh, woh, out);
}
